// Int8WeightLinear_29351806501239
// MI455X (gfx1250) — compile-verified, matrix operand order checked by reading, not run on hardware
//
#include <hip/hip_runtime.h>
#include <stdint.h>

typedef __attribute__((ext_vector_type(16))) _Float16 v16h;
typedef __attribute__((ext_vector_type(8)))  _Float16 v8h;
typedef __attribute__((ext_vector_type(2)))  _Float16 h2;
typedef __attribute__((ext_vector_type(8)))  float    v8f;
typedef __attribute__((ext_vector_type(4)))  float    v4f;
typedef __attribute__((ext_vector_type(4)))  int      v4i;
typedef __attribute__((ext_vector_type(4)))  unsigned v4u;

#define KTOT 4096
#define NTOT 11008
#define BM   128
#define BN   256
#define BK   32
#define NK   (KTOT / BK)      // 128 K-steps
#define LDST 40               // LDS row stride in halves (32 + 8 pad -> 80B rows)
#define PFD  8                // prefetch distance in K-tiles

// Exact int8 -> 2x f16 via magic-number trick:
//   v_perm places byte u into 0x64 pattern -> f16(0x6400|u) == 1024 + u
//   with u = b ^ 0x80 = b + 128, subtracting 1152 yields b exactly.
struct DQ2 { unsigned lo, hi; };

__device__ __forceinline__ DQ2 dq4(unsigned w) {
    const unsigned MAG = 0x64646464u;
    unsigned l = __builtin_amdgcn_perm(MAG, w, 0x04010400u) ^ 0x00800080u;
    unsigned h = __builtin_amdgcn_perm(MAG, w, 0x04030402u) ^ 0x00800080u;
    const h2 bias = { (_Float16)1152.0f, (_Float16)1152.0f };
    DQ2 r;
    r.lo = __builtin_bit_cast(unsigned, (h2)(__builtin_bit_cast(h2, l) - bias));
    r.hi = __builtin_bit_cast(unsigned, (h2)(__builtin_bit_cast(h2, h) - bias));
    return r;
}

__global__ __launch_bounds__(256)
void int8_linear_wmma_f16(const float* __restrict__ A,
                          const int8_t* __restrict__ W,
                          const float* __restrict__ scales,
                          const float* __restrict__ bias,
                          float* __restrict__ out)
{
    // double-buffered tiles: A(activations) 128x32 f16, B(weights) 256x32 f16
    __shared__ _Float16 sA[2][BM * LDST];   // 20 KB
    __shared__ _Float16 sB[2][BN * LDST];   // 40 KB

    const int tid  = threadIdx.x;
    const int lane = tid & 31;
    const int wave = tid >> 5;        // 0..7
    const int wm   = wave >> 2;       // 0..1 : 64-row strip in M
    const int wn   = wave & 3;        // 0..3 : 64-col strip in N
    const int lr   = lane & 15;
    const int lh   = lane >> 4;       // half-wave select per WMMA reg layout

    const int mBase = blockIdx.y * BM;
    const int nBase = blockIdx.x * BN;

    // ---- loader mapping ----
    // A tile 128x32 f32: 2 threads/row, 16 floats each
    const int ldRowA = tid >> 1;          // 0..127
    const int ldColA = (tid & 1) << 4;    // 0 or 16
    // W tile 256x32 i8: 1 thread/row, 32 bytes each
    const float*  gA = A + (size_t)(mBase + ldRowA) * KTOT + ldColA;
    const int8_t* gW = W + (size_t)(nBase + tid) * KTOT;

    v8f acc[4][4];
#pragma unroll
    for (int mt = 0; mt < 4; ++mt)
#pragma unroll
        for (int nt = 0; nt < 4; ++nt)
            acc[mt][nt] = (v8f){};

    // register staging (1 K-tile ahead of LDS)
    v4f ra[4];
    v4i rw[2];

    auto fetch = [&](int kt) {
        const float*  pA = gA + (size_t)kt * BK;
        const int8_t* pW = gW + (size_t)kt * BK;
#pragma unroll
        for (int i = 0; i < 4; ++i) ra[i] = *(const v4f*)(pA + i * 4);
        rw[0] = *(const v4i*)(pW);
        rw[1] = *(const v4i*)(pW + 16);
    };

    auto stage = [&](int buf) {
        // activations: f32 -> f16 (packed cvt)
        {
            _Float16* dst = &sA[buf][ldRowA * LDST + ldColA];
            v8h h0, h1;
#pragma unroll
            for (int i = 0; i < 4; ++i) {
                h0[i]     = (_Float16)ra[0][i];
                h0[i + 4] = (_Float16)ra[1][i];
                h1[i]     = (_Float16)ra[2][i];
                h1[i + 4] = (_Float16)ra[3][i];
            }
            *(v8h*)(dst)     = h0;
            *(v8h*)(dst + 8) = h1;
        }
        // weights: int8 -> f16 via v_perm + v_pk_add_f16 (exact)
        {
            _Float16* dst = &sB[buf][tid * LDST];
#pragma unroll
            for (int q = 0; q < 2; ++q) {       // 16 bytes -> 16 halves each
                v4u p0, p1;
                DQ2 d0 = dq4((unsigned)rw[q][0]);
                DQ2 d1 = dq4((unsigned)rw[q][1]);
                DQ2 d2 = dq4((unsigned)rw[q][2]);
                DQ2 d3 = dq4((unsigned)rw[q][3]);
                p0[0] = d0.lo; p0[1] = d0.hi; p0[2] = d1.lo; p0[3] = d1.hi;
                p1[0] = d2.lo; p1[1] = d2.hi; p1[2] = d3.lo; p1[3] = d3.hi;
                *(v4u*)(dst + q * 16)     = p0;   // k = q*16 .. +7
                *(v4u*)(dst + q * 16 + 8) = p1;   // k = q*16+8 .. +15
            }
        }
    };

    fetch(0);
    stage(0);
    __syncthreads();

    for (int kt = 0; kt < NK; ++kt) {
        const int cur = kt & 1;

        if (kt + 1 < NK) fetch(kt + 1);        // global loads overlap WMMAs
        if (kt + PFD < NK) {                   // global_prefetch_b8 deeper ahead
            __builtin_prefetch(gA + (size_t)(kt + PFD) * BK, 0, 1);
            __builtin_prefetch(gW + (size_t)(kt + PFD) * BK, 0, 1);
        }

        // fragment loads per CDNA5 16-bit A-matrix 16x32 layout:
        // lanes 0-15 : K=0..7 then 16..23 ; lanes 16-31 : K=8..15 then 24..31
        union { v16h v; v8h h[2]; } af[4], bf[4];
#pragma unroll
        for (int mt = 0; mt < 4; ++mt) {
            const _Float16* p =
                &sA[cur][(wm * 64 + mt * 16 + lr) * LDST + lh * 8];
            af[mt].h[0] = *(const v8h*)p;
            af[mt].h[1] = *(const v8h*)(p + 16);
        }
#pragma unroll
        for (int nt = 0; nt < 4; ++nt) {
            const _Float16* p =
                &sB[cur][(wn * 64 + nt * 16 + lr) * LDST + lh * 8];
            bf[nt].h[0] = *(const v8h*)p;
            bf[nt].h[1] = *(const v8h*)(p + 16);
        }

#pragma unroll
        for (int mt = 0; mt < 4; ++mt)
#pragma unroll
            for (int nt = 0; nt < 4; ++nt)
                acc[mt][nt] = __builtin_amdgcn_wmma_f32_16x16x32_f16(
                    false, af[mt].v, false, bf[nt].v,
                    (short)0, acc[mt][nt], false, false);

        if (kt + 1 < NK) stage(cur ^ 1);
        __syncthreads();
    }

    // epilogue: D layout -> VGPR r holds (M=r, N=lane) lanes 0-15,
    // (M=r+8, N=lane-16) lanes 16-31. Fused scale & bias per out channel.
#pragma unroll
    for (int nt = 0; nt < 4; ++nt) {
        const int col = nBase + wn * 64 + nt * 16 + lr;
        const float s = scales[col];
        const float b = bias[col];
#pragma unroll
        for (int mt = 0; mt < 4; ++mt) {
            const int rowBase = mBase + wm * 64 + mt * 16 + lh * 8;
#pragma unroll
            for (int r = 0; r < 8; ++r) {
                out[(size_t)(rowBase + r) * NTOT + col] =
                    acc[mt][nt][r] * s + b;
            }
        }
    }
}

extern "C" void kernel_launch(void* const* d_in, const int* in_sizes, int n_in,
                              void* d_out, int out_size, void* d_ws, size_t ws_size,
                              hipStream_t stream) {
    const float*  A  = (const float*) d_in[0];   // [4,2048,4096] f32
    const int8_t* W  = (const int8_t*)d_in[1];   // [11008,4096] int8
    const float*  S  = (const float*) d_in[2];   // [11008,1]
    const float*  Bi = (const float*) d_in[3];   // [11008]
    float* out = (float*)d_out;                  // [4,2048,11008] f32

    const int M = in_sizes[0] / KTOT;            // 8192 rows
    dim3 grid(NTOT / BN, M / BM);                // (43, 64)
    int8_linear_wmma_f16<<<grid, 256, 0, stream>>>(A, W, S, Bi, out);
}
